// NestedParallelBlock_40862318854319
// MI455X (gfx1250) — matrix operand order repaired and checked by reading, not compiled, not run on hardware
//
#include <hip/hip_runtime.h>
#include <hip/hip_bf16.h>
#include <math.h>

// ---------------- types ----------------
typedef __bf16 bf16_t;
typedef __attribute__((ext_vector_type(16))) __bf16 v16bf;
typedef __attribute__((ext_vector_type(8)))  float  v8f;

#define DIMX      1024
#define NEX       4
#define NHEAD     16
#define HDIM      64
#define EXPAND_O  7168      // 3*DIM + 4*DIM
#define CINX      5120      // DIM + 4*DIM
#define TWO_DIM   2048
#define BT        8192      // B*N tokens
#define OUT_MAIN  8388608   // BT*DIM

static __device__ __forceinline__ bf16_t tobf(float f) { return (bf16_t)f; }

union FragU { v16bf v; uint4 q[2]; };
union H8U   { uint4 u; bf16_t h[8]; };
union H4U   { uint2 u; bf16_t h[4]; };

static __device__ __forceinline__ v8f wmma_bf16(v16bf a, v16bf b, v8f c) {
  return __builtin_amdgcn_wmma_f32_16x16x32_bf16(false, a, false, b, (short)0, c, false, false);
}

// ---------------- fp32 -> bf16 weight conversion ----------------
__global__ __launch_bounds__(256) void cvt_bf16_kernel(const float* __restrict__ s,
                                                       bf16_t* __restrict__ d) {
  long i = ((long)blockIdx.x * 256 + threadIdx.x) * 4;
  float4 f = *(const float4*)(s + i);
  H4U h;
  h.h[0] = tobf(f.x); h.h[1] = tobf(f.y); h.h[2] = tobf(f.z); h.h[3] = tobf(f.w);
  *(uint2*)(d + i) = h.u;
}

// ---------------- LN1 + expert input mask -> bf16 ----------------
__global__ __launch_bounds__(256) void ln_in_kernel(const float* __restrict__ x,
    const int* __restrict__ emask, const float* __restrict__ g,
    const float* __restrict__ b, bf16_t* __restrict__ xn) {
  const int t = blockIdx.x;
  const float* xr = x + (size_t)t * DIMX;
  float4 xv = ((const float4*)xr)[threadIdx.x];
  float s  = xv.x + xv.y + xv.z + xv.w;
  float s2 = xv.x*xv.x + xv.y*xv.y + xv.z*xv.z + xv.w*xv.w;
#pragma unroll
  for (int m = 16; m; m >>= 1) { s += __shfl_xor(s, m); s2 += __shfl_xor(s2, m); }
  __shared__ float ps[8], ps2[8];
  if ((threadIdx.x & 31) == 0) { ps[threadIdx.x >> 5] = s; ps2[threadIdx.x >> 5] = s2; }
  __syncthreads();
  float ts = 0.f, ts2 = 0.f;
#pragma unroll
  for (int i = 0; i < 8; ++i) { ts += ps[i]; ts2 += ps2[i]; }
  const float mean = ts * (1.f / DIMX);
  const float var  = ts2 * (1.f / DIMX) - mean * mean;
  const float inv  = rsqrtf(var + 1e-5f);
  const int shift = (NEX - 1) - emask[t];
  const int din   = DIMX >> shift;
  const int base  = threadIdx.x * 4;
  const float vals[4] = {xv.x, xv.y, xv.z, xv.w};
  H4U o;
#pragma unroll
  for (int j = 0; j < 4; ++j) {
    int c = base + j;
    float v = ((vals[j] - mean) * inv * g[c] + b[c]) * (c < din ? 1.f : 0.f);
    o.h[j] = tobf(v);
  }
  *(uint2*)(xn + (size_t)t * DIMX + base) = o.u;
}

// ---------------- LN2 over k and v halves of kv buffer -> bf16 ----------------
__global__ __launch_bounds__(256) void ln_kv_kernel(const float* __restrict__ kv,
    const float* __restrict__ g, const float* __restrict__ b,
    bf16_t* __restrict__ kout, bf16_t* __restrict__ vout) {
  const int t = blockIdx.x >> 1, which = blockIdx.x & 1;
  const float* src = kv + (size_t)t * TWO_DIM + which * DIMX;
  float4 xv = ((const float4*)src)[threadIdx.x];
  float s  = xv.x + xv.y + xv.z + xv.w;
  float s2 = xv.x*xv.x + xv.y*xv.y + xv.z*xv.z + xv.w*xv.w;
#pragma unroll
  for (int m = 16; m; m >>= 1) { s += __shfl_xor(s, m); s2 += __shfl_xor(s2, m); }
  __shared__ float ps[8], ps2[8];
  if ((threadIdx.x & 31) == 0) { ps[threadIdx.x >> 5] = s; ps2[threadIdx.x >> 5] = s2; }
  __syncthreads();
  float ts = 0.f, ts2 = 0.f;
#pragma unroll
  for (int i = 0; i < 8; ++i) { ts += ps[i]; ts2 += ps2[i]; }
  const float mean = ts * (1.f / DIMX);
  const float var  = ts2 * (1.f / DIMX) - mean * mean;
  const float inv  = rsqrtf(var + 1e-5f);
  const int base = threadIdx.x * 4;
  const float vals[4] = {xv.x, xv.y, xv.z, xv.w};
  H4U o;
#pragma unroll
  for (int j = 0; j < 4; ++j) {
    int c = base + j;
    o.h[j] = tobf((vals[j] - mean) * inv * g[c] + b[c]);
  }
  bf16_t* dst = (which ? vout : kout) + (size_t)t * DIMX;
  *(uint2*)(dst + base) = o.u;
}

// ---------------- TN GEMM: C[m,n] = sum_k A[m,k]*B[n,k], bf16 in, f32 acc ----
// Block tile 128x256, 8 waves as 2(M) x 4(N), wave tile 64x64 (16 WMMAs/K-step).
// MODE 0: contract (Z = C + cbias, f32, ldc = 2048)
// MODE 1: expand   (route cols: [0,1024)->q bf16, [1024,3072)->kv f32,
//                   [3072,7168)->gelu(+mlp_bias)-> h bf16 at col 1024+cc)
template <int MODE>
__global__ __launch_bounds__(256) void gemm_tn_kernel(
    const bf16_t* __restrict__ A, const bf16_t* __restrict__ Bw, int K,
    float* __restrict__ Z, const float* __restrict__ cbias,
    bf16_t* __restrict__ qout, float* __restrict__ kvout,
    bf16_t* __restrict__ hout, const float* __restrict__ mlpb) {
  __shared__ bf16_t sA[2][128][40];
  __shared__ bf16_t sB[2][256][40];
  const int tid = threadIdx.x;
  const int wid = tid >> 5, lane = tid & 31;
  const int lhalf = lane >> 4, l16 = lane & 15;
  const int bm = blockIdx.y * 128, bn = blockIdx.x * 256;
  const int waveM = wid & 1, waveN = wid >> 1;   // 2 (M) x 4 (N) wave grid
  const int larow = tid >> 1;                    // 0..127  (A staging row)
  const int lak = (tid & 1) * 16;                // 0 or 16 (A staging k)
  const bf16_t* gA = A + (size_t)(bm + larow) * K + lak;
  const bf16_t* gB = Bw + (size_t)(bn + tid) * K;   // one B row per thread
  uint4 ra0 = *(const uint4*)gA, ra1 = *(const uint4*)(gA + 8);
  uint4 rb0 = *(const uint4*)gB,        rb1 = *(const uint4*)(gB + 8);
  uint4 rb2 = *(const uint4*)(gB + 16), rb3 = *(const uint4*)(gB + 24);
  const v8f vzero = {0.f,0.f,0.f,0.f,0.f,0.f,0.f,0.f};
  v8f acc[4][4];
#pragma unroll
  for (int mt = 0; mt < 4; ++mt)
#pragma unroll
    for (int nt = 0; nt < 4; ++nt) acc[mt][nt] = vzero;
  const int ks = K >> 5;
  for (int kt = 0; kt < ks; ++kt) {
    const int buf = kt & 1;
    *(uint4*)&sA[buf][larow][lak]     = ra0;
    *(uint4*)&sA[buf][larow][lak + 8] = ra1;
    *(uint4*)&sB[buf][tid][0]  = rb0;
    *(uint4*)&sB[buf][tid][8]  = rb1;
    *(uint4*)&sB[buf][tid][16] = rb2;
    *(uint4*)&sB[buf][tid][24] = rb3;
    if (kt + 1 < ks) {
      gA += 32; gB += 32;
      ra0 = *(const uint4*)gA;        ra1 = *(const uint4*)(gA + 8);
      rb0 = *(const uint4*)gB;        rb1 = *(const uint4*)(gB + 8);
      rb2 = *(const uint4*)(gB + 16); rb3 = *(const uint4*)(gB + 24);
      // warm GL2 one K-tile ahead of the register preload
      __builtin_prefetch(gA + 32, 0, 2);
      __builtin_prefetch(gB + 32, 0, 2);
    }
    __syncthreads();
    FragU a[4], b[4];
#pragma unroll
    for (int mt = 0; mt < 4; ++mt) {
      const bf16_t* p = &sA[buf][waveM * 64 + mt * 16 + l16][lhalf * 8];
      a[mt].q[0] = *(const uint4*)p;
      a[mt].q[1] = *(const uint4*)(p + 16);
    }
#pragma unroll
    for (int nt = 0; nt < 4; ++nt) {
      const bf16_t* p = &sB[buf][waveN * 64 + nt * 16 + l16][lhalf * 8];
      b[nt].q[0] = *(const uint4*)p;
      b[nt].q[1] = *(const uint4*)(p + 16);
    }
#pragma unroll
    for (int mt = 0; mt < 4; ++mt)
#pragma unroll
      for (int nt = 0; nt < 4; ++nt)
        acc[mt][nt] = wmma_bf16(a[mt].v, b[nt].v, acc[mt][nt]);
    __syncthreads();
  }
  // epilogue: lane holds col = l16, rows v + 8*lhalf of each 16x16 tile
#pragma unroll
  for (int nt = 0; nt < 4; ++nt) {
    const int col = bn + waveN * 64 + nt * 16 + l16;
#pragma unroll
    for (int mt = 0; mt < 4; ++mt) {
      const int row0 = bm + waveM * 64 + mt * 16 + lhalf * 8;
      v8f c = acc[mt][nt];
      if (MODE == 0) {
        const float bias = cbias[col];
#pragma unroll
        for (int v = 0; v < 8; ++v)
          Z[(size_t)(row0 + v) * TWO_DIM + col] = c[v] + bias;
      } else {
        if (col < DIMX) {
#pragma unroll
          for (int v = 0; v < 8; ++v)
            qout[(size_t)(row0 + v) * DIMX + col] = tobf(c[v]);
        } else if (col < 3 * DIMX) {
          const int cc = col - DIMX;
#pragma unroll
          for (int v = 0; v < 8; ++v)
            kvout[(size_t)(row0 + v) * TWO_DIM + cc] = c[v];
        } else {
          const int cc = col - 3 * DIMX;
          const float mb = mlpb[cc];
#pragma unroll
          for (int v = 0; v < 8; ++v) {
            float gx = c[v] + mb;
            gx = 0.5f * gx * (1.f + erff(gx * 0.70710678118654752f));
            hout[(size_t)(row0 + v) * CINX + DIMX + cc] = tobf(gx);
          }
        }
      }
    }
  }
}

// ---------------- flash attention (per b, head, 128-query block) ----------
__global__ __launch_bounds__(256) void attn_kernel(
    const bf16_t* __restrict__ Q, const bf16_t* __restrict__ Km,
    const bf16_t* __restrict__ Vm, bf16_t* __restrict__ H) {
  __shared__ bf16_t sK[2][32][72];    // [key][d], row 144B
  __shared__ bf16_t sVt[2][64][40];   // [d][key], row 80B
  __shared__ bf16_t sP[8][16][40];    // per-wave P tile 16x32
  const int b = blockIdx.z, hh = blockIdx.y, qb = blockIdx.x;
  const int tid = threadIdx.x, wid = tid >> 5, lane = tid & 31;
  const int lhalf = lane >> 4, l16 = lane & 15;
  const size_t tokbase = (size_t)b * 1024;
  const int qrow = qb * 128 + wid * 16;
  // Q A-fragments (d = 0..31, 32..63), kept in registers for all key blocks
  FragU aq[2];
  const bf16_t* qp = Q + (tokbase + qrow + l16) * DIMX + hh * HDIM;
#pragma unroll
  for (int f = 0; f < 2; ++f) {
    const int d0 = f * 32 + lhalf * 8;
    aq[f].q[0] = *(const uint4*)(qp + d0);
    aq[f].q[1] = *(const uint4*)(qp + d0 + 16);
  }
  const v8f vzero = {0.f,0.f,0.f,0.f,0.f,0.f,0.f,0.f};
  v8f o[4];
#pragma unroll
  for (int nt = 0; nt < 4; ++nt) o[nt] = vzero;
  float mrow[8], lrow[8];
#pragma unroll
  for (int v = 0; v < 8; ++v) { mrow[v] = -INFINITY; lrow[v] = 0.f; }
  const int skey = tid >> 3;          // 0..31
  const int sd   = (tid & 7) * 8;     // 0..56
  const float scale = 0.125f;         // HD^-0.5
  for (int kb = 0; kb < 32; ++kb) {
    const int buf = kb & 1;
    const size_t krow = tokbase + kb * 32 + skey;
    H8U kvr; kvr.u = *(const uint4*)(Km + krow * DIMX + hh * HDIM + sd);
    *(uint4*)&sK[buf][skey][sd] = kvr.u;
    H8U vvr; vvr.u = *(const uint4*)(Vm + krow * DIMX + hh * HDIM + sd);
#pragma unroll
    for (int j = 0; j < 8; ++j) sVt[buf][sd + j][skey] = vvr.h[j];  // transpose
    __syncthreads();
    // S = Q K^T  (two 16-key column tiles, K-dim = 64 -> 2 wmma each)
    FragU bk[2][2];
#pragma unroll
    for (int jt = 0; jt < 2; ++jt)
#pragma unroll
      for (int df = 0; df < 2; ++df) {
        const bf16_t* p = &sK[buf][jt * 16 + l16][df * 32 + lhalf * 8];
        bk[jt][df].q[0] = *(const uint4*)p;
        bk[jt][df].q[1] = *(const uint4*)(p + 16);
      }
    v8f s[2]; s[0] = vzero; s[1] = vzero;
#pragma unroll
    for (int jt = 0; jt < 2; ++jt)
#pragma unroll
      for (int df = 0; df < 2; ++df)
        s[jt] = wmma_bf16(aq[df].v, bk[jt][df].v, s[jt]);
    // online softmax update, row stats live in the C-fragment row mapping
#pragma unroll
    for (int v = 0; v < 8; ++v) {
      float s0 = s[0][v] * scale, s1 = s[1][v] * scale;
      float mx = fmaxf(s0, s1);
#pragma unroll
      for (int m = 8; m; m >>= 1) mx = fmaxf(mx, __shfl_xor(mx, m));
      const float mnew = fmaxf(mrow[v], mx);
      const float al   = __expf(mrow[v] - mnew);
      const float p0 = __expf(s0 - mnew), p1 = __expf(s1 - mnew);
      float rs = p0 + p1;
#pragma unroll
      for (int m = 8; m; m >>= 1) rs += __shfl_xor(rs, m);
      lrow[v] = lrow[v] * al + rs;
      mrow[v] = mnew;
#pragma unroll
      for (int nt = 0; nt < 4; ++nt) o[nt][v] *= al;
      sP[wid][v + lhalf * 8][l16]      = tobf(p0);
      sP[wid][v + lhalf * 8][16 + l16] = tobf(p1);
    }
    // O += P @ V  (K-dim = 32 keys, 4 output d-tiles)
    FragU ap;
    const bf16_t* pp = &sP[wid][l16][lhalf * 8];
    ap.q[0] = *(const uint4*)pp;
    ap.q[1] = *(const uint4*)(pp + 16);
    FragU bv[4];
#pragma unroll
    for (int nt = 0; nt < 4; ++nt) {
      const bf16_t* vp = &sVt[buf][nt * 16 + l16][lhalf * 8];
      bv[nt].q[0] = *(const uint4*)vp;
      bv[nt].q[1] = *(const uint4*)(vp + 16);
    }
#pragma unroll
    for (int nt = 0; nt < 4; ++nt)
      o[nt] = wmma_bf16(ap.v, bv[nt].v, o[nt]);
    __syncthreads();
  }
#pragma unroll
  for (int v = 0; v < 8; ++v) {
    const float inv = 1.f / lrow[v];
    const size_t trow = tokbase + qrow + v + lhalf * 8;
#pragma unroll
    for (int nt = 0; nt < 4; ++nt)
      H[trow * CINX + hh * HDIM + nt * 16 + l16] = tobf(o[nt][v] * inv);
  }
}

// ---------------- final epilogue: masks, residual, router-weighted MLP ----
__global__ __launch_bounds__(256) void final_kernel(
    const float* __restrict__ Z, const float* __restrict__ x,
    const int* __restrict__ emask, const float* __restrict__ rp,
    const float* __restrict__ alpha, float* __restrict__ out) {
  const int t = blockIdx.x;
  const int e = emask[t];
  const int dout = TWO_DIM >> (3 - e);
  const float w = alpha[0] * rp[t * NEX + e] + 1.0f;
  const int c = threadIdx.x * 4;
  const float* z = Z + (size_t)t * TWO_DIM;
  float4 za = *(const float4*)(z + c);
  float4 zb = *(const float4*)(z + DIMX + c);
  float4 xv = *(const float4*)(x + (size_t)t * DIMX + c);
  float4 r;
  r.x = (c + 0 < dout ? za.x : 0.f) + xv.x + w * ((DIMX + c + 0) < dout ? zb.x : 0.f);
  r.y = (c + 1 < dout ? za.y : 0.f) + xv.y + w * ((DIMX + c + 1) < dout ? zb.y : 0.f);
  r.z = (c + 2 < dout ? za.z : 0.f) + xv.z + w * ((DIMX + c + 2) < dout ? zb.z : 0.f);
  r.w = (c + 3 < dout ? za.w : 0.f) + xv.w + w * ((DIMX + c + 3) < dout ? zb.w : 0.f);
  *(float4*)(out + (size_t)t * DIMX + c) = r;
}

// ---------------- tuple tail: expert_mask bits + router_probs ----------------
__global__ __launch_bounds__(256) void tail_kernel(const int* __restrict__ emask,
                                                   const float* __restrict__ rp,
                                                   float* __restrict__ out) {
  const int i = blockIdx.x * 256 + threadIdx.x;
  if (i < BT) ((int*)(out + OUT_MAIN))[i] = emask[i];
  if (i < BT * NEX) out[OUT_MAIN + BT + i] = rp[i];
}

// ---------------- launch ----------------
extern "C" void kernel_launch(void* const* d_in, const int* in_sizes, int n_in,
                              void* d_out, int out_size, void* d_ws, size_t ws_size,
                              hipStream_t stream) {
  (void)in_sizes; (void)n_in; (void)out_size; (void)ws_size;
  const float* x     = (const float*)d_in[0];
  const int*   emask = (const int*)d_in[1];
  const float* rp    = (const float*)d_in[2];
  const float* wexp  = (const float*)d_in[3];
  const float* mlpb  = (const float*)d_in[4];
  const float* wcon  = (const float*)d_in[5];
  const float* cbias = (const float*)d_in[6];
  const float* n1g   = (const float*)d_in[7];
  const float* n1b   = (const float*)d_in[8];
  const float* n2g   = (const float*)d_in[9];
  const float* n2b   = (const float*)d_in[10];
  const float* alpha = (const float*)d_in[11];
  float* out = (float*)d_out;

  char* ws = (char*)d_ws;
  size_t off = 0;
  auto alloc = [&](size_t bytes) {
    void* p = ws + off;
    off += (bytes + 255) & ~(size_t)255;
    return p;
  };
  bf16_t* xn_bf   = (bf16_t*)alloc((size_t)BT * DIMX * 2);
  bf16_t* wexp_bf = (bf16_t*)alloc((size_t)EXPAND_O * DIMX * 2);
  bf16_t* wcon_bf = (bf16_t*)alloc((size_t)TWO_DIM * CINX * 2);
  bf16_t* q_bf    = (bf16_t*)alloc((size_t)BT * DIMX * 2);
  float*  kv_f    = (float*)alloc((size_t)BT * TWO_DIM * 4);
  bf16_t* k_bf    = (bf16_t*)alloc((size_t)BT * DIMX * 2);
  bf16_t* v_bf    = (bf16_t*)alloc((size_t)BT * DIMX * 2);
  bf16_t* h_bf    = (bf16_t*)alloc((size_t)BT * CINX * 2);
  float*  z_f     = (float*)alloc((size_t)BT * TWO_DIM * 4);

  cvt_bf16_kernel<<<(EXPAND_O * DIMX) / 1024, 256, 0, stream>>>(wexp, wexp_bf);
  cvt_bf16_kernel<<<(TWO_DIM * CINX) / 1024, 256, 0, stream>>>(wcon, wcon_bf);
  ln_in_kernel<<<BT, 256, 0, stream>>>(x, emask, n1g, n1b, xn_bf);
  gemm_tn_kernel<1><<<dim3(EXPAND_O / 256, BT / 128), 256, 0, stream>>>(
      xn_bf, wexp_bf, DIMX, nullptr, nullptr, q_bf, kv_f, h_bf, mlpb);
  ln_kv_kernel<<<BT * 2, 256, 0, stream>>>(kv_f, n2g, n2b, k_bf, v_bf);
  attn_kernel<<<dim3(8, NHEAD, 8), 256, 0, stream>>>(q_bf, k_bf, v_bf, h_bf);
  gemm_tn_kernel<0><<<dim3(TWO_DIM / 256, BT / 128), 256, 0, stream>>>(
      h_bf, wcon_bf, CINX, z_f, cbias, nullptr, nullptr, nullptr, nullptr);
  final_kernel<<<BT, 256, 0, stream>>>(z_f, x, emask, rp, alpha, out);
  tail_kernel<<<(BT * NEX + 255) / 256, 256, 0, stream>>>(emask, rp, out);
}
